// SoftCodebook_74895639708074
// MI455X (gfx1250) — matrix operand order repaired and checked by reading, not compiled, not run on hardware
//
#include <hip/hip_runtime.h>
#include <hip/hip_bf16.h>

typedef _Float16 h8  __attribute__((ext_vector_type(8)));
typedef _Float16 h16 __attribute__((ext_vector_type(16)));
typedef float    f4  __attribute__((ext_vector_type(4)));
typedef float    f8  __attribute__((ext_vector_type(8)));

#define K_PROTO 128
#define D_DIM   256
#define M_TILE  128

// LDS row strides in f16 elements (padded to de-conflict the 64 banks)
#define PS  264   // sProto  [128][PS]  (normalized c_bar, [k][d])
#define TS  136   // sProtoT [256][TS]  (raw prototypes, [d][k])
#define HS  264   // sH      [128][HS]  (normalized h_bar, [m][d])
#define SPS 136   // sP      [128][SPS] (softmax probs f16, [m][k])

static constexpr int OFF_PROTO  = 0;
static constexpr int OFF_PROTOT = OFF_PROTO  + K_PROTO * PS;
static constexpr int OFF_H      = OFF_PROTOT + D_DIM   * TS;
static constexpr int OFF_P      = OFF_H      + M_TILE  * HS;
static constexpr int SMEM_HALFS = OFF_P      + M_TILE  * SPS;
static constexpr size_t SMEM_BYTES = (size_t)SMEM_HALFS * 2;  // ~234 KB < 320 KB/WGP

__device__ __forceinline__ h16 ld16(const _Float16* p) {
    h8 lo = *(const h8*)(p);
    h8 hi = *(const h8*)(p + 16);
    h16 r;
#pragma unroll
    for (int i = 0; i < 8; ++i) { r[i] = lo[i]; r[8 + i] = hi[i]; }
    return r;
}

__global__ void __launch_bounds__(256)
soft_codebook_fused(const float* __restrict__ g_h,
                    const float* __restrict__ g_proto,
                    float* __restrict__ out_p,
                    float* __restrict__ out_z)
{
    extern __shared__ __align__(16) _Float16 smem[];
    _Float16* sProto  = smem + OFF_PROTO;
    _Float16* sProtoT = smem + OFF_PROTOT;
    _Float16* sH      = smem + OFF_H;
    _Float16* sP      = smem + OFF_P;

    const int tid    = threadIdx.x;
    const int wave   = tid >> 5;     // 8 waves of 32 (wave32)
    const int lane   = tid & 31;
    const int kh     = lane >> 4;    // which half-wave (K/M split in fragments)
    const int nl     = lane & 15;
    const int R0     = blockIdx.x * M_TILE;

    // ---------------- Phase A1: prototypes -> LDS ----------------
    // sProto  = l2norm(proto) as f16 (B operand of sim GEMM)
    // sProtoT = raw proto transposed as f16 (B operand of z GEMM)
    for (int r = wave; r < K_PROTO; r += 8) {
        const float* src = g_proto + (size_t)r * D_DIM + lane * 8;
        f4 a = *(const f4*)(src);
        f4 b = *(const f4*)(src + 4);
        float v[8] = { a[0], a[1], a[2], a[3], b[0], b[1], b[2], b[3] };
        float ssq = 0.f;
#pragma unroll
        for (int i = 0; i < 8; ++i) ssq += v[i] * v[i];
#pragma unroll
        for (int off = 16; off >= 1; off >>= 1) ssq += __shfl_xor(ssq, off, 32);
        const float scale = 1.0f / fmaxf(sqrtf(ssq), 1e-12f);
        h8 nrm;
#pragma unroll
        for (int i = 0; i < 8; ++i) nrm[i] = (_Float16)(v[i] * scale);
        *(h8*)(sProto + r * PS + lane * 8) = nrm;
#pragma unroll
        for (int i = 0; i < 8; ++i)
            sProtoT[(lane * 8 + i) * TS + r] = (_Float16)v[i];
    }

    // ---------------- Phase A2: h tile -> normalized f16 in sH ----------------
    for (int r = wave; r < M_TILE; r += 8) {
        const float* src = g_h + (size_t)(R0 + r) * D_DIM + lane * 8;
        if (r + 8 < M_TILE)
            __builtin_prefetch(g_h + (size_t)(R0 + r + 8) * D_DIM + lane * 8, 0, 0);
        f4 a = *(const f4*)(src);
        f4 b = *(const f4*)(src + 4);
        float v[8] = { a[0], a[1], a[2], a[3], b[0], b[1], b[2], b[3] };
        float ssq = 0.f;
#pragma unroll
        for (int i = 0; i < 8; ++i) ssq += v[i] * v[i];
#pragma unroll
        for (int off = 16; off >= 1; off >>= 1) ssq += __shfl_xor(ssq, off, 32);
        const float scale = 1.0f / fmaxf(sqrtf(ssq), 1e-12f);
        h8 o;
#pragma unroll
        for (int i = 0; i < 8; ++i) o[i] = (_Float16)(v[i] * scale);
        *(h8*)(sH + r * HS + lane * 8) = o;
    }
    __syncthreads();

    // ---------------- Phase B: sim = h_bar @ c_bar^T (16 rows per wave) ----------------
    const int m0 = wave * 16;
    const int mA = m0 + nl;                 // A-fragment row for this lane
    f8 acc[8];
#pragma unroll
    for (int nt = 0; nt < 8; ++nt)
#pragma unroll
        for (int j = 0; j < 8; ++j) acc[nt][j] = 0.0f;

#pragma unroll
    for (int t = 0; t < 8; ++t) {           // K-dim (d) steps of 32
        const int dbase = t * 32;
        const int kb = dbase + kh * 8;      // 16-bit A layout: lanes>=16 hold K+8
        h16 afrag = [&]() {
            h8 lo = *(const h8*)(sH + mA * HS + kb);
            h8 hi = *(const h8*)(sH + mA * HS + kb + 16);
            h16 r;
#pragma unroll
            for (int i = 0; i < 8; ++i) { r[i] = lo[i]; r[8 + i] = hi[i]; }
            return r;
        }();
        const int dB = dbase + kh * 8;      // B layout: lanes>=16 hold K+8 (same as A)
#pragma unroll
        for (int nt = 0; nt < 8; ++nt) {
            const int N = nt * 16 + nl;     // prototype index (column of sim)
            h16 bfrag = ld16(sProto + N * PS + dB);
            acc[nt] = __builtin_amdgcn_wmma_f32_16x16x32_f16(
                false, afrag, false, bfrag, (short)0, acc[nt], false, false);
        }
    }

    // ---------------- Phase C: softmax over K=128, emit p (f32 -> HBM, f16 -> LDS) ----------------
    // C/D layout: lane (kh,nl), VGPR j -> element [m0 + kh*8 + j, nt*16 + nl]
#pragma unroll
    for (int j = 0; j < 8; ++j) {
        const int row = m0 + kh * 8 + j;
        float v[8];
        float mx = -3.402823466e38f;
#pragma unroll
        for (int nt = 0; nt < 8; ++nt) { v[nt] = acc[nt][j] * 10.0f; mx = fmaxf(mx, v[nt]); }
#pragma unroll
        for (int off = 8; off >= 1; off >>= 1) mx = fmaxf(mx, __shfl_xor(mx, off, 16));
        float s = 0.f;
#pragma unroll
        for (int nt = 0; nt < 8; ++nt) { v[nt] = __expf(v[nt] - mx); s += v[nt]; }
#pragma unroll
        for (int off = 8; off >= 1; off >>= 1) s += __shfl_xor(s, off, 16);
        const float inv = 1.0f / s;
        float* prow = out_p + (size_t)(R0 + row) * K_PROTO;
#pragma unroll
        for (int nt = 0; nt < 8; ++nt) {
            const float pv = v[nt] * inv;
            prow[nt * 16 + nl] = pv;
            sP[row * SPS + nt * 16 + nl] = (_Float16)pv;
        }
    }
    __syncthreads();

    // ---------------- Phase D: z = p @ proto (N = 256 in two chunks of 128) ----------------
#pragma unroll
    for (int chunk = 0; chunk < 2; ++chunk) {
        f8 zacc[8];
#pragma unroll
        for (int nt = 0; nt < 8; ++nt)
#pragma unroll
            for (int j = 0; j < 8; ++j) zacc[nt][j] = 0.0f;

#pragma unroll
        for (int t = 0; t < 4; ++t) {       // K-dim (proto index) steps of 32
            const int kb = t * 32 + kh * 8;
            h8 lo = *(const h8*)(sP + mA * SPS + kb);
            h8 hi = *(const h8*)(sP + mA * SPS + kb + 16);
            h16 afrag;
#pragma unroll
            for (int i = 0; i < 8; ++i) { afrag[i] = lo[i]; afrag[8 + i] = hi[i]; }
            const int koff = t * 32 + kh * 8;
#pragma unroll
            for (int nt = 0; nt < 8; ++nt) {
                const int dcol = chunk * 128 + nt * 16 + nl;   // output dim index
                h16 bfrag = ld16(sProtoT + dcol * TS + koff);
                zacc[nt] = __builtin_amdgcn_wmma_f32_16x16x32_f16(
                    false, afrag, false, bfrag, (short)0, zacc[nt], false, false);
            }
        }
#pragma unroll
        for (int nt = 0; nt < 8; ++nt) {
#pragma unroll
            for (int j = 0; j < 8; ++j) {
                const int row = m0 + kh * 8 + j;
                out_z[(size_t)(R0 + row) * D_DIM + chunk * 128 + nt * 16 + nl] = zacc[nt][j];
            }
        }
    }
}

extern "C" void kernel_launch(void* const* d_in, const int* in_sizes, int n_in,
                              void* d_out, int out_size, void* d_ws, size_t ws_size,
                              hipStream_t stream) {
    const float* g_h     = (const float*)d_in[0];   // (B,P,D) f32, B*P*D elems
    const float* g_proto = (const float*)d_in[1];   // (K,D)   f32
    const int nRows = in_sizes[0] / D_DIM;          // B*P = 65536

    float* out_p = (float*)d_out;                          // (B,P,K) f32
    float* out_z = out_p + (size_t)nRows * K_PROTO;        // (B,P,D) f32

    (void)n_in; (void)out_size; (void)d_ws; (void)ws_size;

    hipFuncSetAttribute(reinterpret_cast<const void*>(soft_codebook_fused),
                        hipFuncAttributeMaxDynamicSharedMemorySize, (int)SMEM_BYTES);

    dim3 grid(nRows / M_TILE);   // 512 workgroups
    dim3 block(256);             // 8 wave32 waves
    soft_codebook_fused<<<grid, block, SMEM_BYTES, stream>>>(g_h, g_proto, out_p, out_z);
}
